// GraphSAGENet_38079180046954
// MI455X (gfx1250) — matrix operand order repaired and checked by reading, not compiled, not run on hardware
//
#include <hip/hip_runtime.h>
#include <hip/hip_bf16.h>

#define D 128

typedef __attribute__((ext_vector_type(16))) __bf16 v16bf;
typedef __attribute__((ext_vector_type(8)))  float  v8f;

// ---------------------------------------------------------------------------
// Build the WMMA A-operand (16x32 bf16, contiguous-k order, same as B) for one K-chunk of 32.
// Lane L (L<16) holds row M=L, K = kbase+{0..15}; lanes 16..31 hold
// K = kbase+{16..31}.  `row` already points at this lane's A row.
// ---------------------------------------------------------------------------
__device__ __forceinline__ v16bf load_a_chunk(const float* __restrict__ row,
                                              int kbase, int hf, float scale) {
  const float* p = row + kbase + (hf << 4);
  float4 a0 = *reinterpret_cast<const float4*>(p);
  float4 a1 = *reinterpret_cast<const float4*>(p + 4);
  float4 c0 = *reinterpret_cast<const float4*>(p + 8);
  float4 c1 = *reinterpret_cast<const float4*>(p + 12);
  v16bf v;
  v[0]  = (__bf16)(a0.x * scale); v[1]  = (__bf16)(a0.y * scale);
  v[2]  = (__bf16)(a0.z * scale); v[3]  = (__bf16)(a0.w * scale);
  v[4]  = (__bf16)(a1.x * scale); v[5]  = (__bf16)(a1.y * scale);
  v[6]  = (__bf16)(a1.z * scale); v[7]  = (__bf16)(a1.w * scale);
  v[8]  = (__bf16)(c0.x * scale); v[9]  = (__bf16)(c0.y * scale);
  v[10] = (__bf16)(c0.z * scale); v[11] = (__bf16)(c0.w * scale);
  v[12] = (__bf16)(c1.x * scale); v[13] = (__bf16)(c1.y * scale);
  v[14] = (__bf16)(c1.z * scale); v[15] = (__bf16)(c1.w * scale);
  return v;
}

// B-operand (32x16 bf16): lane n<16 needs W[j=16*nt+n][k16grp*16 .. +15],
// lanes 16..31 read the next k16 group.  LDS layout [kgroup][j][16] makes the
// 16 bf16 contiguous -> two ds_load_b128 per lane.
__device__ __forceinline__ v16bf load_b16(const __bf16* __restrict__ base,
                                          int kgroup, int j) {
  union { float4 f4[2]; v16bf v; } u;
  const float4* p =
      reinterpret_cast<const float4*>(base + (((kgroup << 7) + j) << 4));
  u.f4[0] = p[0];
  u.f4[1] = p[1];
  return u.v;
}

// ---------------------------------------------------------------------------
// Edge kernels: degree count, inverse degree, mean-aggregation scatter.
// ---------------------------------------------------------------------------
__global__ __launch_bounds__(256) void degree_kernel(
    const int* __restrict__ dst, float* __restrict__ deg, int e) {
  int i = blockIdx.x * blockDim.x + threadIdx.x;
  int stride = gridDim.x * blockDim.x;
  for (; i < e; i += stride) unsafeAtomicAdd(&deg[dst[i]], 1.0f);
}

__global__ __launch_bounds__(256) void invdeg_kernel(float* __restrict__ deg, int n) {
  int i = blockIdx.x * blockDim.x + threadIdx.x;
  if (i < n) {
    float d = deg[i];
    deg[i] = d > 0.0f ? 1.0f / d : 0.0f;
  }
}

// One wave per edge: lane l moves features [4l, 4l+4) with a float4 load and
// four hardware fp32 atomics (global_atomic_add_f32).
__global__ __launch_bounds__(256) void aggregate_kernel(
    const float* __restrict__ h, const int* __restrict__ src,
    const int* __restrict__ dst, float* __restrict__ agg, int e) {
  int wid  = (blockIdx.x * blockDim.x + threadIdx.x) >> 5;
  int lane = threadIdx.x & 31;
  int nw   = (gridDim.x * blockDim.x) >> 5;
  for (int ed = wid; ed < e; ed += nw) {
    int s = src[ed];
    int d = dst[ed];
    float4 v = *reinterpret_cast<const float4*>(h + (size_t)s * D + lane * 4);
    float* o = agg + (size_t)d * D + lane * 4;
    unsafeAtomicAdd(o + 0, v.x);
    unsafeAtomicAdd(o + 1, v.y);
    unsafeAtomicAdd(o + 2, v.z);
    unsafeAtomicAdd(o + 3, v.w);
  }
}

// ---------------------------------------------------------------------------
// Fused SAGE layer: out = SiLU(LN( agg*invdeg @ Wl^T + bl + h @ Wr^T + br ))
// One wave computes a 16-node x 128-feature tile via bf16 WMMA (K=256 concat).
// ---------------------------------------------------------------------------
__global__ __launch_bounds__(256) void sage_layer_kernel(
    const float* __restrict__ agg, const float* __restrict__ hin,
    const float* __restrict__ invdeg,
    const float* __restrict__ Wl, const float* __restrict__ Wr,
    const float* __restrict__ bl, const float* __restrict__ br,
    const float* __restrict__ g,  const float* __restrict__ bln,
    float* __restrict__ out, int n) {
  __shared__ __align__(16) __bf16 Wlds[2][8][D][16];  // 64 KB: Wl, Wr as bf16

  // Stage both weight matrices into LDS, converted to bf16, in B layout.
  for (int idx = threadIdx.x; idx < 2 * D * D; idx += blockDim.x) {
    int wsel = idx >> 14;
    int r    = idx & 16383;
    int j    = r >> 7;
    int k    = r & 127;
    const float* Wsrc = wsel ? Wr : Wl;
    Wlds[wsel][k >> 4][j][k & 15] = (__bf16)Wsrc[j * D + k];
  }
  __syncthreads();

  int lane = threadIdx.x & 31;
  int hf   = lane >> 4;
  int l15  = lane & 15;
  int wave = threadIdx.x >> 5;
  int tile = blockIdx.x * 8 + wave;
  int ntiles = (n + 15) >> 4;
  if (tile >= ntiles) return;
  int base = tile << 4;

  int mrow = base + l15;
  if (mrow >= n) mrow = n - 1;
  float sc = invdeg[mrow];
  const float* arow = agg + (size_t)mrow * D;
  const float* hrow = hin + (size_t)mrow * D;

  // All A operands up front (makes the in-place layer-2 store safe).
  v16bf A[8];
#pragma unroll
  for (int c = 0; c < 4; ++c) A[c]     = load_a_chunk(arow, 32 * c, hf, sc);
#pragma unroll
  for (int c = 0; c < 4; ++c) A[4 + c] = load_a_chunk(hrow, 32 * c, hf, 1.0f);

  v8f acc[8];
  float gv[8], bv[8];
#pragma unroll
  for (int nt = 0; nt < 8; ++nt) {
    int f = (nt << 4) + l15;
    float cinit = bl[f] + br[f];
    gv[nt] = g[f];
    bv[nt] = bln[f];
    v8f cc;
#pragma unroll
    for (int r = 0; r < 8; ++r) cc[r] = cinit;
#pragma unroll
    for (int c = 0; c < 8; ++c) {
      int wsel = c >> 2;                    // 0: Wl with agg, 1: Wr with h
      int kg   = ((c & 3) << 1) | hf;
      v16bf B = load_b16(&Wlds[wsel][0][0][0], kg, f);
      cc = __builtin_amdgcn_wmma_f32_16x16x32_bf16(
          false, A[c], false, B, (short)0, cc, false, false);
    }
    acc[nt] = cc;
  }

  // LayerNorm (half-wave shfl reduction) + SiLU, store 16x128 tile.
#pragma unroll
  for (int r = 0; r < 8; ++r) {
    float s1 = 0.0f, s2 = 0.0f;
#pragma unroll
    for (int nt = 0; nt < 8; ++nt) {
      float v = acc[nt][r];
      s1 += v;
      s2 += v * v;
    }
#pragma unroll
    for (int m = 1; m <= 8; m <<= 1) {
      s1 += __shfl_xor(s1, m, 32);
      s2 += __shfl_xor(s2, m, 32);
    }
    float mu  = s1 * (1.0f / D);
    float var = s2 * (1.0f / D) - mu * mu;
    float rs  = rsqrtf(var + 1e-5f);
    int orow = base + (hf << 3) + r;
    if (orow < n) {
      float* op = out + (size_t)orow * D;
#pragma unroll
      for (int nt = 0; nt < 8; ++nt) {
        float v = (acc[nt][r] - mu) * rs * gv[nt] + bv[nt];
        op[(nt << 4) + l15] = v * (1.0f / (1.0f + __expf(-v)));  // SiLU
      }
    }
  }
}

// ---------------------------------------------------------------------------
// MLP head: out = relu(h @ w1^T + b1) @ w2^T + b2   (128 -> 128 -> 1)
// ---------------------------------------------------------------------------
__global__ __launch_bounds__(256) void mlp_head_kernel(
    const float* __restrict__ hin, const float* __restrict__ w1,
    const float* __restrict__ b1,  const float* __restrict__ w2,
    const float* __restrict__ b2,  float* __restrict__ out, int n) {
  __shared__ __align__(16) __bf16 W1lds[8][D][16];  // 32 KB
  __shared__ float w2lds[D];
  __shared__ float b1lds[D];

  for (int idx = threadIdx.x; idx < D * D; idx += blockDim.x) {
    int j = idx >> 7;
    int k = idx & 127;
    W1lds[k >> 4][j][k & 15] = (__bf16)w1[j * D + k];
  }
  if (threadIdx.x < D) {
    w2lds[threadIdx.x] = w2[threadIdx.x];
    b1lds[threadIdx.x] = b1[threadIdx.x];
  }
  __syncthreads();

  int lane = threadIdx.x & 31;
  int hf   = lane >> 4;
  int l15  = lane & 15;
  int wave = threadIdx.x >> 5;
  int tile = blockIdx.x * 8 + wave;
  int ntiles = (n + 15) >> 4;
  if (tile >= ntiles) return;
  int base = tile << 4;

  int mrow = base + l15;
  if (mrow >= n) mrow = n - 1;
  const float* hrow = hin + (size_t)mrow * D;

  v16bf A[4];
#pragma unroll
  for (int c = 0; c < 4; ++c) A[c] = load_a_chunk(hrow, 32 * c, hf, 1.0f);

  float part[8];
#pragma unroll
  for (int r = 0; r < 8; ++r) part[r] = 0.0f;

#pragma unroll
  for (int nt = 0; nt < 8; ++nt) {
    int f = (nt << 4) + l15;
    float cinit = b1lds[f];
    v8f cc;
#pragma unroll
    for (int r = 0; r < 8; ++r) cc[r] = cinit;
#pragma unroll
    for (int c = 0; c < 4; ++c) {
      int kg = (c << 1) | hf;
      v16bf B = load_b16(&W1lds[0][0][0], kg, f);
      cc = __builtin_amdgcn_wmma_f32_16x16x32_bf16(
          false, A[c], false, B, (short)0, cc, false, false);
    }
    float wv = w2lds[f];
#pragma unroll
    for (int r = 0; r < 8; ++r) part[r] += fmaxf(cc[r], 0.0f) * wv;
  }

  float bb = b2[0];
#pragma unroll
  for (int r = 0; r < 8; ++r) {
    float s = part[r];
#pragma unroll
    for (int m = 1; m <= 8; m <<= 1) s += __shfl_xor(s, m, 32);
    int orow = base + (hf << 3) + r;
    if (l15 == 0 && orow < n) out[orow] = s + bb;
  }
}

// ---------------------------------------------------------------------------
extern "C" void kernel_launch(void* const* d_in, const int* in_sizes, int n_in,
                              void* d_out, int out_size, void* d_ws, size_t ws_size,
                              hipStream_t stream) {
  const float* x   = (const float*)d_in[0];
  const int*   ei  = (const int*)d_in[1];   // [2, E] int32
  const float* llw = (const float*)d_in[2];
  const float* llb = (const float*)d_in[3];
  const float* lrw = (const float*)d_in[4];
  const float* lrb = (const float*)d_in[5];
  const float* lng = (const float*)d_in[6];
  const float* lnb = (const float*)d_in[7];
  const float* w1  = (const float*)d_in[8];
  const float* b1  = (const float*)d_in[9];
  const float* w2  = (const float*)d_in[10];
  const float* b2  = (const float*)d_in[11];

  const int n = in_sizes[0] / D;
  const int e = in_sizes[1] / 2;
  const int* srcI = ei;
  const int* dstI = ei + e;

  char* ws = (char*)d_ws;
  size_t degBytes = (((size_t)n * 4) + 255) & ~(size_t)255;
  float* deg  = (float*)ws;                    // N (becomes inv_deg)
  float* agg  = (float*)(ws + degBytes);       // N*D
  float* hbuf = agg + (size_t)n * D;           // N*D

  hipMemsetAsync(deg, 0, (size_t)n * 4, stream);
  degree_kernel<<<1024, 256, 0, stream>>>(dstI, deg, e);
  invdeg_kernel<<<(n + 255) / 256, 256, 0, stream>>>(deg, n);

  int ntiles  = (n + 15) / 16;
  int gblocks = (ntiles + 7) / 8;

  // Layer 0: aggregate from x, fused GEMM+LN+SiLU -> hbuf
  hipMemsetAsync(agg, 0, (size_t)n * D * 4, stream);
  aggregate_kernel<<<4096, 256, 0, stream>>>(x, srcI, dstI, agg, e);
  sage_layer_kernel<<<gblocks, 256, 0, stream>>>(agg, x, deg, llw, lrw, llb,
                                                 lrb, lng, lnb, hbuf, n);

  // Layer 1: aggregate from hbuf, write hbuf in place (per-wave safe)
  hipMemsetAsync(agg, 0, (size_t)n * D * 4, stream);
  aggregate_kernel<<<4096, 256, 0, stream>>>(hbuf, srcI, dstI, agg, e);
  sage_layer_kernel<<<gblocks, 256, 0, stream>>>(
      agg, hbuf, deg, llw + D * D, lrw + D * D, llb + D, lrb + D, lng + D,
      lnb + D, hbuf, n);

  // MLP head -> d_out [N]
  mlp_head_kernel<<<gblocks, 256, 0, stream>>>(hbuf, w1, b1, w2, b2,
                                               (float*)d_out, n);
}
